// AxialTransformerBlock_16183436771854
// MI455X (gfx1250) — matrix operand order repaired and checked by reading, not compiled, not run on hardware
//
#include <hip/hip_runtime.h>
#include <hip/hip_bf16.h>
#include <math.h>

typedef __attribute__((ext_vector_type(16))) __bf16 v16bf;
typedef __attribute__((ext_vector_type(8)))  float  v8f;

union FragBF { v16bf v; uint4 q[2]; };

#define EMBED 384
#define TOKS  768   // 3*16*16 tokens per batch element

// Async DMA global -> LDS, 16B per lane, tracked by ASYNCcnt (gfx1250).
__device__ __forceinline__ void async_ld_b128(unsigned lds_off, const void* gptr) {
  asm volatile("global_load_async_to_lds_b128 %0, %1, off"
               :: "v"(lds_off), "v"(gptr) : "memory");
}
__device__ __forceinline__ void wait_async_le4() {
  asm volatile("s_wait_asynccnt 0x4" ::: "memory");
}
__device__ __forceinline__ void wait_async_0() {
  asm volatile("s_wait_asynccnt 0x0" ::: "memory");
}

// ---------------------------------------------------------------------------
// Weight conversion / packing kernels (fp32 -> bf16)
// ---------------------------------------------------------------------------
__global__ void pack_qkv_bf16(const float* __restrict__ Wq,
                              const float* __restrict__ Wkv,
                              __bf16* __restrict__ dst) {
  int idx = blockIdx.x * 256 + threadIdx.x;           // 384 x 1152
  if (idx >= 384 * 1152) return;
  int r = idx / 1152, c = idx % 1152;
  float v = (c < 384) ? Wq[r * 384 + c] : Wkv[r * 768 + (c - 384)];
  dst[idx] = (__bf16)v;
}

__global__ void cat_wo_bf16(const float* __restrict__ W0,
                            const float* __restrict__ W1,
                            const float* __restrict__ W2,
                            __bf16* __restrict__ dst) {
  int idx = blockIdx.x * 256 + threadIdx.x;           // 1152 x 384
  if (idx >= 1152 * 384) return;
  int r = idx / 384, c = idx % 384;
  const float* W = (r < 384) ? W0 : (r < 768) ? W1 : W2;
  dst[idx] = (__bf16)W[(r % 384) * 384 + c];
}

__global__ void cvt_bf16(const float* __restrict__ src,
                         __bf16* __restrict__ dst, int n) {
  int idx = blockIdx.x * 256 + threadIdx.x;
  if (idx < n) dst[idx] = (__bf16)src[idx];
}

__global__ void bias_sum3(const float* __restrict__ a, const float* __restrict__ b,
                          const float* __restrict__ c, float* __restrict__ d) {
  int i = blockIdx.x * 256 + threadIdx.x;
  if (i < 384) d[i] = a[i] + b[i] + c[i];
}

// ---------------------------------------------------------------------------
// Tiled transposes: x (B,384,768) <-> token-major (B,768,384)
// ---------------------------------------------------------------------------
__global__ __launch_bounds__(256)
void transpose_in(const float* __restrict__ x, float* __restrict__ Xf,
                  __bf16* __restrict__ Xb) {
  __shared__ float tile[32][33];
  int b = blockIdx.z;
  int p0 = blockIdx.x * 32;          // token index within batch elem
  int c0 = blockIdx.y * 32;          // channel
  int tx = threadIdx.x & 31, ty = threadIdx.x >> 5;
  const float* xb = x + (size_t)b * EMBED * TOKS;
#pragma unroll
  for (int i = 0; i < 4; ++i) {
    int c = c0 + ty + i * 8;
    tile[ty + i * 8][tx] = xb[(size_t)c * TOKS + p0 + tx];   // coalesced over p
  }
  __syncthreads();
  float*  Xfb = Xf + (size_t)b * TOKS * EMBED;
  __bf16* Xbb = Xb + (size_t)b * TOKS * EMBED;
#pragma unroll
  for (int i = 0; i < 4; ++i) {
    int p = p0 + ty + i * 8;
    float v = tile[tx][ty + i * 8];
    Xfb[(size_t)p * EMBED + c0 + tx] = v;                    // coalesced over c
    Xbb[(size_t)p * EMBED + c0 + tx] = (__bf16)v;
  }
}

__global__ __launch_bounds__(256)
void transpose_out(const float* __restrict__ t2, float* __restrict__ out) {
  __shared__ float tile[32][33];
  int b = blockIdx.z;
  int c0 = blockIdx.x * 32;
  int p0 = blockIdx.y * 32;
  int tx = threadIdx.x & 31, ty = threadIdx.x >> 5;
  const float* tb = t2 + (size_t)b * TOKS * EMBED;
#pragma unroll
  for (int i = 0; i < 4; ++i) {
    int p = p0 + ty + i * 8;
    tile[ty + i * 8][tx] = tb[(size_t)p * EMBED + c0 + tx];  // coalesced over c
  }
  __syncthreads();
  float* ob = out + (size_t)b * EMBED * TOKS;
#pragma unroll
  for (int i = 0; i < 4; ++i) {
    int c = c0 + ty + i * 8;
    ob[(size_t)c * TOKS + p0 + tx] = tile[tx][ty + i * 8];   // coalesced over p
  }
}

// ---------------------------------------------------------------------------
// WMMA bf16 GEMM: C(MxN) = A(MxK, row-major bf16) @ B(KxN, row-major bf16)
// Double-buffered LDS filled by global_load_async_to_lds_b128 (ASYNCcnt).
// Software-pipelined: nk = K/32 is even for every GEMM in this model, so the
// steady loop handles two tiles with compile-time buffer indices.
// MODE 0: store bf16          MODE 1: += bias[col] + res -> store f32
// MODE 2: += bias[col], exact GELU -> store bf16
// Block tile 128x128, 8 waves, each wave 32(M) x 64(N) via 2x4 wmma tiles.
// ---------------------------------------------------------------------------
template <int MODE>
__global__ __launch_bounds__(256)
void gemm_bf16_wmma(const __bf16* __restrict__ A, const __bf16* __restrict__ B,
                    int K, int N,
                    const float* __restrict__ bias,
                    const float* __restrict__ res, int ldr,
                    void* __restrict__ Cout) {
  __shared__ __align__(16) __bf16 As[2][128][40];   // 128 x 32 (+8 pad) x2 buf
  __shared__ __align__(16) __bf16 Bs[2][32][136];   // 32 x 128 (+8 pad) x2 buf

  const int tid  = threadIdx.x;
  const int lane = tid & 31;
  const int wave = tid >> 5;
  const int wr = wave >> 1;                 // 0..3 -> M offset 32*wr
  const int wc = wave & 1;                  // 0..1 -> N offset 64*wc
  const size_t m0 = (size_t)blockIdx.y * 128;
  const int    n0 = blockIdx.x * 128;

  // Per-thread tile chunks: A chunks (tid, tid+256) -> rows (tid>>2, +64);
  // B chunks (tid, tid+256) -> rows (tid>>4, +16).
  const int rA = tid >> 2, cA = (tid & 3) * 8;
  const int rB = tid >> 4, cB = (tid & 15) * 8;
  const __bf16* gA0 = A + (m0 + rA) * (size_t)K + cA;
  const __bf16* gA1 = gA0 + (size_t)64 * K;
  const __bf16* gB0 = B + (size_t)rB * N + n0 + cB;
  const __bf16* gB1 = gB0 + (size_t)16 * N;
  const unsigned aOff0 = (unsigned)(uintptr_t)&As[0][rA][cA];
  const unsigned aOff1 = (unsigned)(uintptr_t)&As[0][rA + 64][cA];
  const unsigned bOff0 = (unsigned)(uintptr_t)&Bs[0][rB][cB];
  const unsigned bOff1 = (unsigned)(uintptr_t)&Bs[0][rB + 16][cB];
  const unsigned aStride = 128 * 40 * sizeof(__bf16);   // buffer 1 delta
  const unsigned bStride = 32 * 136 * sizeof(__bf16);
  const size_t gBstep = (size_t)32 * N;

  auto issue_tile = [&](int buf) {
    unsigned ab = buf ? aStride : 0u, bb = buf ? bStride : 0u;
    async_ld_b128(aOff0 + ab, gA0);
    async_ld_b128(aOff1 + ab, gA1);
    async_ld_b128(bOff0 + bb, gB0);
    async_ld_b128(bOff1 + bb, gB1);
    gA0 += 32; gA1 += 32; gB0 += gBstep; gB1 += gBstep;
  };

  const v8f vz = {0.f, 0.f, 0.f, 0.f, 0.f, 0.f, 0.f, 0.f};
  v8f acc[2][4];
#pragma unroll
  for (int i = 0; i < 2; ++i)
#pragma unroll
    for (int j = 0; j < 4; ++j) acc[i][j] = vz;

  auto compute_tile = [&](int cur) {       // cur is a compile-time constant at
    FragBF a[2], bf[4];                    // every call site below
    const int m = lane & 15, half = lane >> 4;
#pragma unroll
    for (int ms = 0; ms < 2; ++ms) {       // A: lane=M, e0..7 -> K half*8..,
      int rowi = wr * 32 + ms * 16 + m;    //    e8..15 -> 16+half*8..
      a[ms].q[0] = *(const uint4*)&As[cur][rowi][half * 8];
      a[ms].q[1] = *(const uint4*)&As[cur][rowi][16 + half * 8];
    }
#pragma unroll
    for (int ns = 0; ns < 4; ++ns) {       // B: lane=N, e0..7 -> K half*8.., e8..15 -> 16+half*8..
      int nOff = wc * 64 + ns * 16;
#pragma unroll
      for (int s = 0; s < 8; ++s) {
        bf[ns].v[s]     = Bs[cur][half * 8 + s][nOff + m];
        bf[ns].v[8 + s] = Bs[cur][16 + half * 8 + s][nOff + m];
      }
    }
#pragma unroll
    for (int ms = 0; ms < 2; ++ms)
#pragma unroll
      for (int ns = 0; ns < 4; ++ns)
        acc[ms][ns] = __builtin_amdgcn_wmma_f32_16x16x32_bf16(
            false, a[ms].v, false, bf[ns].v, (short)0, acc[ms][ns], false, false);
  };

  const int nk = K >> 5;                   // always even (12 / 36 / 48)
  issue_tile(0);                           // prologue: tile 0 -> buf 0

  for (int kt = 0; kt < nk - 2; kt += 2) {
    issue_tile(1); wait_async_le4();
    __syncthreads();
    compute_tile(0);
    __syncthreads();
    issue_tile(0); wait_async_le4();
    __syncthreads();
    compute_tile(1);
    __syncthreads();
  }
  issue_tile(1); wait_async_le4();         // tile nk-1 -> buf1; tile nk-2 ready
  __syncthreads();
  compute_tile(0);
  __syncthreads();
  wait_async_0();
  __syncthreads();
  compute_tile(1);                         // last tile; no trailing barrier

  // Epilogue. C layout: lane 0-15 N=lane M=r; lane 16-31 N=lane-16 M=r+8.
  const int half = lane >> 4;
  const int nlane = lane & 15;
#pragma unroll
  for (int ms = 0; ms < 2; ++ms) {
#pragma unroll
    for (int ns = 0; ns < 4; ++ns) {
      int col = n0 + wc * 64 + ns * 16 + nlane;
#pragma unroll
      for (int r = 0; r < 8; ++r) {
        size_t row = m0 + wr * 32 + ms * 16 + half * 8 + r;
        float v = acc[ms][ns][r];
        if (MODE == 0) {
          ((__bf16*)Cout)[row * (size_t)N + col] = (__bf16)v;
        } else if (MODE == 1) {
          v += bias[col] + res[row * (size_t)ldr + col];
          ((float*)Cout)[row * (size_t)N + col] = v;
        } else {
          v += bias[col];
          v = 0.5f * v * (1.0f + erff(v * 0.70710678118654752f));   // exact GELU
          ((__bf16*)Cout)[row * (size_t)N + col] = (__bf16)v;
        }
      }
    }
  }
}

// ---------------------------------------------------------------------------
// Axial attention: one wave per (sequence, head). T in {3,16}, DH=32.
// qkv: (Ntok x 1152) bf16 [q|k|v];  oc: (Ntok x 1152) bf16, col block = axis*384
// ---------------------------------------------------------------------------
__global__ __launch_bounds__(256)
void axial_attn(const __bf16* __restrict__ qkv, __bf16* __restrict__ oc, int axis) {
  __shared__ float sm[8][3][16][32];     // per-wave q,k,v tiles (fp32)
  const int lane = threadIdx.x & 31;
  const int w    = threadIdx.x >> 5;
  const int task = blockIdx.x * 8 + w;
  const int seq  = task / 12;
  const int head = task % 12;

  int base, stride, T;
  if (axis == 0) {                       // attend along s (T=3), stride 256
    int b = seq >> 8; int hw = seq & 255;
    base = b * 768 + hw; stride = 256; T = 3;
  } else {
    int b = seq / 48; int j = seq % 48; int s = j >> 4; int r = j & 15;
    if (axis == 1) { base = b * 768 + s * 256 + r;      stride = 16; }  // along h
    else           { base = b * 768 + s * 256 + r * 16; stride = 1;  }  // along w
    T = 16;
  }

  for (int idx = lane; idx < T * 32; idx += 32) {
    int t = idx >> 5, d = idx & 31;
    const __bf16* p = qkv + (size_t)(base + t * stride) * 1152 + head * 32 + d;
    sm[w][0][t][d] = (float)p[0];
    sm[w][1][t][d] = (float)p[384];
    sm[w][2][t][d] = (float)p[768];
  }
  __syncthreads();

  if (lane < T) {
    float qreg[32];
#pragma unroll
    for (int d = 0; d < 32; ++d) qreg[d] = sm[w][0][lane][d];
    float sc[16];
    float mx = -1e30f;
    for (int j = 0; j < T; ++j) {
      float dot = 0.f;
#pragma unroll
      for (int d = 0; d < 32; ++d) dot += qreg[d] * sm[w][1][j][d];
      dot *= 0.17677669529663688f;       // 32^-0.5
      sc[j] = dot;
      mx = fmaxf(mx, dot);
    }
    float sum = 0.f;
    for (int j = 0; j < T; ++j) { sc[j] = __expf(sc[j] - mx); sum += sc[j]; }
    float inv = 1.f / sum;
#pragma unroll 4
    for (int d = 0; d < 32; ++d) {
      float o = 0.f;
      for (int j = 0; j < T; ++j) o += sc[j] * sm[w][2][j][d];
      sm[w][0][lane][d] = o * inv;       // reuse q tile for output
    }
  }
  __syncthreads();

  for (int idx = lane; idx < T * 32; idx += 32) {
    int t = idx >> 5, d = idx & 31;
    oc[(size_t)(base + t * stride) * 1152 + axis * 384 + head * 32 + d] =
        (__bf16)sm[w][0][t][d];
  }
}

// ---------------------------------------------------------------------------
// LayerNorm over 384 channels; one wave per row (12 elems/lane, shuffle reduce)
// ---------------------------------------------------------------------------
__device__ __forceinline__ float wave_sum(float v) {
  for (int off = 16; off > 0; off >>= 1) v += __shfl_xor(v, off, 32);
  return v;
}

__global__ __launch_bounds__(256)
void layernorm384(const float* __restrict__ in, const float* __restrict__ g,
                  const float* __restrict__ be, float* __restrict__ of,
                  __bf16* __restrict__ ob) {
  const int lane = threadIdx.x & 31;
  const int w    = threadIdx.x >> 5;
  const size_t row = (size_t)blockIdx.x * 8 + w;
  const float* rp = in + row * 384;
  float e[12];
  float s = 0.f;
#pragma unroll
  for (int i = 0; i < 12; ++i) { e[i] = rp[lane + i * 32]; s += e[i]; }
  s = wave_sum(s);
  float mean = s * (1.0f / 384.0f);
  float vs = 0.f;
#pragma unroll
  for (int i = 0; i < 12; ++i) { float d = e[i] - mean; vs += d * d; }
  vs = wave_sum(vs);
  float rs = rsqrtf(vs * (1.0f / 384.0f) + 1e-5f);
#pragma unroll
  for (int i = 0; i < 12; ++i) {
    int c = lane + i * 32;
    float v = (e[i] - mean) * rs * g[c] + be[c];
    if (of) of[row * 384 + c] = v;
    if (ob) ob[row * 384 + c] = (__bf16)v;
  }
}

// ---------------------------------------------------------------------------
extern "C" void kernel_launch(void* const* d_in, const int* in_sizes, int n_in,
                              void* d_out, int out_size, void* d_ws, size_t ws_size,
                              hipStream_t stream) {
  const float* x = (const float*)d_in[0];
  const float* Wq[3]  = {(const float*)d_in[1], (const float*)d_in[5], (const float*)d_in[9]};
  const float* Wkv[3] = {(const float*)d_in[2], (const float*)d_in[6], (const float*)d_in[10]};
  const float* Wo[3]  = {(const float*)d_in[3], (const float*)d_in[7], (const float*)d_in[11]};
  const float* bo[3]  = {(const float*)d_in[4], (const float*)d_in[8], (const float*)d_in[12]};
  const float* g1  = (const float*)d_in[13];
  const float* be1 = (const float*)d_in[14];
  const float* W1m = (const float*)d_in[15];
  const float* b1m = (const float*)d_in[16];
  const float* W2m = (const float*)d_in[17];
  const float* b2m = (const float*)d_in[18];
  const float* g2  = (const float*)d_in[19];
  const float* be2 = (const float*)d_in[20];

  const int Bsz = in_sizes[0] / (EMBED * TOKS);
  const size_t Ntok = (size_t)Bsz * TOKS;

  size_t off = 0;
  char* ws = (char*)d_ws;
  auto wsAlloc = [&](size_t bytes) -> void* {
    void* p = ws + off;
    off += (bytes + 255) & ~(size_t)255;
    return p;
  };
  __bf16* wqkv[3];
  for (int i = 0; i < 3; ++i) wqkv[i] = (__bf16*)wsAlloc(384 * 1152 * sizeof(__bf16));
  __bf16* wocat = (__bf16*)wsAlloc(1152 * 384 * sizeof(__bf16));
  __bf16* w1    = (__bf16*)wsAlloc(384 * 1536 * sizeof(__bf16));
  __bf16* w2    = (__bf16*)wsAlloc(1536 * 384 * sizeof(__bf16));
  float*  bosum = (float*)wsAlloc(384 * sizeof(float));
  float*  Xf    = (float*)wsAlloc(Ntok * 384 * sizeof(float));   // token-major x, later LN2 out
  __bf16* Xb    = (__bf16*)wsAlloc(Ntok * 384 * sizeof(__bf16)); // x bf16, later t bf16
  float*  Dt    = (float*)wsAlloc(Ntok * 384 * sizeof(float));   // t_pre / t (post-LN1)
  __bf16* Abuf  = (__bf16*)wsAlloc(Ntok * 1536 * sizeof(__bf16)); // QKV, later MLP hidden
  void*   Bbuf  = wsAlloc(Ntok * 1152 * sizeof(__bf16));          // Oc bf16, later t2_pre f32

  // 1) pack weights to bf16
  for (int i = 0; i < 3; ++i)
    pack_qkv_bf16<<<(384 * 1152 + 255) / 256, 256, 0, stream>>>(Wq[i], Wkv[i], wqkv[i]);
  cat_wo_bf16<<<(1152 * 384 + 255) / 256, 256, 0, stream>>>(Wo[0], Wo[1], Wo[2], wocat);
  cvt_bf16<<<(384 * 1536 + 255) / 256, 256, 0, stream>>>(W1m, w1, 384 * 1536);
  cvt_bf16<<<(1536 * 384 + 255) / 256, 256, 0, stream>>>(W2m, w2, 1536 * 384);
  bias_sum3<<<2, 256, 0, stream>>>(bo[0], bo[1], bo[2], bosum);

  // 2) x -> token-major fp32 + bf16
  transpose_in<<<dim3(TOKS / 32, EMBED / 32, Bsz), 256, 0, stream>>>(x, Xf, Xb);

  const int Mb = (int)(Ntok / 128);     // M tiles

  // 3) per-axial: QKV GEMM then attention into concatenated O buffer
  for (int i = 0; i < 3; ++i) {
    gemm_bf16_wmma<0><<<dim3(1152 / 128, Mb), 256, 0, stream>>>(
        Xb, wqkv[i], 384, 1152, nullptr, nullptr, 0, Abuf);
    int blocks = (i == 0) ? (Bsz * 256 * 12 / 8) : (Bsz * 48 * 12 / 8);
    axial_attn<<<blocks, 256, 0, stream>>>(Abuf, (__bf16*)Bbuf, i);
  }

  // 4) summed output projection + bias + residual -> t_pre (f32)
  gemm_bf16_wmma<1><<<dim3(384 / 128, Mb), 256, 0, stream>>>(
      (const __bf16*)Bbuf, wocat, 1152, 384, bosum, Xf, 384, Dt);

  // 5) LN1 -> t (f32 in place) + t bf16
  layernorm384<<<(int)(Ntok / 8), 256, 0, stream>>>(Dt, g1, be1, Dt, Xb);

  // 6) MLP GEMM1 + GELU -> hidden bf16
  gemm_bf16_wmma<2><<<dim3(1536 / 128, Mb), 256, 0, stream>>>(
      Xb, w1, 384, 1536, b1m, nullptr, 0, Abuf);

  // 7) MLP GEMM2 + bias + residual(t) -> t2_pre f32
  gemm_bf16_wmma<1><<<dim3(384 / 128, Mb), 256, 0, stream>>>(
      Abuf, w2, 1536, 384, b2m, Dt, 384, (float*)Bbuf);

  // 8) LN2 -> token-major f32
  layernorm384<<<(int)(Ntok / 8), 256, 0, stream>>>(
      (const float*)Bbuf, g2, be2, Xf, nullptr);

  // 9) back to (B, C, 3,16,16)
  transpose_out<<<dim3(EMBED / 32, TOKS / 32, Bsz), 256, 0, stream>>>(Xf, (float*)d_out);
}
